// PhysMamba_SSSD_22479858827562
// MI455X (gfx1250) — matrix operand order repaired and checked by reading, not compiled, not run on hardware
//
#include <hip/hip_runtime.h>
#include <hip/hip_bf16.h>
#include <math.h>

typedef __attribute__((ext_vector_type(16))) _Float16 v16h;
typedef __attribute__((ext_vector_type(8)))  float    v8f;
typedef int i32x4 __attribute__((ext_vector_type(4)));

#define BSZ     16
#define LSEQ    1024
#define BL      (BSZ * LSEQ)          // 16384 rows
#define DMODEL  128
#define DINNER  256
#define DSTATE  16
#define DTRANK  8
#define NHEADS  4
#define HD      32
#define FREQ    513                   // L/2 + 1
#define FREQP   544                   // padded to multiple of 32 (fast-path K)

#define ACT_NONE     0
#define ACT_SILU     1
#define ACT_SIGMOID  2
#define ACT_SOFTPLUS 3
#define ACT_GELU     4
#define ACT_RELU     5

// ---- gfx1250 async-copy availability probes --------------------------------
#if defined(__has_builtin)
#if __has_builtin(__builtin_amdgcn_global_load_async_to_lds_b128)
#define HAVE_GLB_ASYNC 1
#else
#define HAVE_GLB_ASYNC 0
#endif
#if __has_builtin(__builtin_amdgcn_s_wait_asynccnt)
#define HAVE_WAIT_ASYNC 1
#else
#define HAVE_WAIT_ASYNC 0
#endif
#else
#define HAVE_GLB_ASYNC 0
#define HAVE_WAIT_ASYNC 0
#endif

#if HAVE_GLB_ASYNC
typedef __attribute__((address_space(1))) i32x4* glb_v4p;   // prints as 'i32x4 __device__ *'
typedef __attribute__((address_space(3))) i32x4* lds_v4p;   // prints as 'i32x4 __shared__ *'
#endif

template <int N>
__device__ __forceinline__ void wait_async() {
#if HAVE_WAIT_ASYNC
  __builtin_amdgcn_s_wait_asynccnt(N);
#else
  asm volatile("s_wait_asynccnt %0" :: "i"(N) : "memory");
#endif
}

// ---------------------------------------------------------------- device utils

__device__ __forceinline__ float dev_act(float v, int act) {
  switch (act) {
    case ACT_SILU:     return v / (1.f + __expf(-v));
    case ACT_SIGMOID:  return 1.f / (1.f + __expf(-v));
    case ACT_SOFTPLUS: return (v > 20.f) ? v : log1pf(__expf(v));
    case ACT_GELU:     return 0.5f * v * (1.f + erff(v * 0.70710678f));
    case ACT_RELU:     return fmaxf(v, 0.f);
    default:           return v;
  }
}

__device__ __forceinline__ void pack4(v16h& d, int base, float4 v) {
  d[base + 0] = (_Float16)v.x;
  d[base + 1] = (_Float16)v.y;
  d[base + 2] = (_Float16)v.z;
  d[base + 3] = (_Float16)v.w;
}

// ---- fast (unguarded, b128) fragment loaders -------------------------------
// A fragment (16x32): lane = M row; elements 0-15 -> K=kb+0..15,
// kb = 0 (lanes 0-15) / 16 (lanes 16-31).
__device__ __forceinline__ v16h load_a_v4(const float* __restrict__ A, int lda,
                                          int m0, int k0, int lane) {
  const float* p = A + (long)(m0 + (lane & 15)) * lda + k0 + ((lane < 16) ? 0 : 16);
  v16h a;
  pack4(a, 0,  *(const float4*)(p));
  pack4(a, 4,  *(const float4*)(p + 4));
  pack4(a, 8,  *(const float4*)(p + 8));
  pack4(a, 12, *(const float4*)(p + 12));
  return a;
}

// B fragment (32x16) from W[N,K]: lane = N col; lanes 0-15 K=k0+0..15, 16-31 K=k0+16..31.
__device__ __forceinline__ v16h load_bT_v4(const float* __restrict__ W, int ldw,
                                           int n0, int k0, int lane) {
  const float* p = W + (long)(n0 + (lane & 15)) * ldw + k0 + ((lane < 16) ? 0 : 16);
  v16h b;
  pack4(b, 0,  *(const float4*)(p));
  pack4(b, 4,  *(const float4*)(p + 4));
  pack4(b, 8,  *(const float4*)(p + 8));
  pack4(b, 12, *(const float4*)(p + 12));
  return b;
}

// B fragment from [K,N] row-major (strided, unguarded).
__device__ __forceinline__ v16h load_bN_fast(const float* __restrict__ Bm, int ldb,
                                             int n0, int k0, int lane) {
  int n  = n0 + (lane & 15);
  int kb = k0 + ((lane < 16) ? 0 : 16);
  v16h b;
#pragma unroll
  for (int i = 0; i < 16; ++i) b[i] = (_Float16)Bm[(long)(kb + i) * ldb + n];
  return b;
}

#if HAVE_GLB_ASYNC
// Stage a 16x32-f32 A tile into LDS (row-major [16][32]) with async b128 copies.
// 128 16-byte chunks; 4 instructions x 32 lanes.
__device__ __forceinline__ void issue_a_async(const float* A, int lda, int m0, int k0,
                                              int lane, float* dst) {
#pragma unroll
  for (int i = 0; i < 4; ++i) {
    int chunk = i * 32 + lane;          // 0..127
    int row   = chunk >> 3;             // 0..15
    int c4    = (chunk & 7) << 2;       // float offset within row
    const float* g = A + (long)(m0 + row) * lda + k0 + c4;
    float* l = dst + row * 32 + c4;
    __builtin_amdgcn_global_load_async_to_lds_b128(
        (glb_v4p)(void*)g, (lds_v4p)(void*)l, 0, 0);
  }
}

__device__ __forceinline__ v16h read_a_lds(const float* As, int lane) {
  const float* p = As + (lane & 15) * 32 + ((lane < 16) ? 0 : 16);
  v16h a;
  pack4(a, 0,  *(const float4*)(p));
  pack4(a, 4,  *(const float4*)(p + 4));
  pack4(a, 8,  *(const float4*)(p + 8));
  pack4(a, 12, *(const float4*)(p + 12));
  return a;
}
#endif

// ---- guarded loaders: clamp address (always in-range load) + value select ---
__device__ __forceinline__ v16h load_a_g(const float* __restrict__ A, int lda,
                                         int M, int K, int m0, int k0, int lane) {
  int m  = m0 + (lane & 15);
  int mc = (m < M) ? m : (M - 1);
  int kb = (lane < 16) ? 0 : 16;
  v16h a;
#pragma unroll
  for (int i = 0; i < 16; ++i) {
    int k  = k0 + kb + i;
    int kc = (k < K) ? k : (K - 1);
    float v = A[(long)mc * lda + kc];
    a[i] = (_Float16)((m < M && k < K) ? v : 0.f);
  }
  return a;
}

__device__ __forceinline__ v16h load_bT_g(const float* __restrict__ W, int ldw,
                                          int N, int K, int n0, int k0, int lane) {
  int n  = n0 + (lane & 15);
  int nc = (n < N) ? n : (N - 1);
  int kb = (lane < 16) ? 0 : 16;
  v16h b;
#pragma unroll
  for (int i = 0; i < 16; ++i) {
    int k  = k0 + kb + i;
    int kc = (k < K) ? k : (K - 1);
    float v = W[(long)nc * ldw + kc];
    b[i] = (_Float16)((n < N && k < K) ? v : 0.f);
  }
  return b;
}

__device__ __forceinline__ v16h load_bN_g(const float* __restrict__ Bm, int ldb,
                                          int N, int K, int n0, int k0, int lane) {
  int n  = n0 + (lane & 15);
  int nc = (n < N) ? n : (N - 1);
  int kb = (lane < 16) ? 0 : 16;
  v16h b;
#pragma unroll
  for (int i = 0; i < 16; ++i) {
    int k  = k0 + kb + i;
    int kc = (k < K) ? k : (K - 1);
    float v = Bm[(long)kc * ldb + nc];
    b[i] = (_Float16)((n < N && k < K) ? v : 0.f);
  }
  return b;
}

// ---------------------------------------------------------------- WMMA GEMM
// C[m,n] = act( sum_k A[m,k]*B[k,n] (+bias[n]) (+C_old) ); B from W[N,K] (transB) or [K,N].
// One wave computes a 16x32 C tile: one A fragment feeds two WMMAs.
// Fast transB path double-buffers the A tile through LDS via async copies.
__global__ void __launch_bounds__(256)
gemm_wmma_k(const float* __restrict__ A, const float* __restrict__ W,
            const float* __restrict__ bias, float* __restrict__ C,
            int M, int N, int K, int lda, int ldw, int ldc,
            long sA, long sW, long sC, int transB, int accum, int act) {
#if HAVE_GLB_ASYNC
  __shared__ float Asm[8][2][16 * 32];
#endif
  A += (long)blockIdx.y * sA;
  W += (long)blockIdx.y * sW;
  C += (long)blockIdx.y * sC;
  int lane = threadIdx.x & 31;
  int wid  = threadIdx.x >> 5;
  int wave = blockIdx.x * (blockDim.x >> 5) + wid;
  int ntn = (N + 31) >> 5;
  int ntm = (M + 15) >> 4;
  if (wave >= ntn * ntm) return;            // wave-uniform: EXEC all-ones for WMMA
  int m0 = (wave / ntn) << 4;
  int n0 = (wave % ntn) << 5;
  bool alA   = ((lda & 3) == 0);
  bool alW   = ((ldw & 3) == 0);
  bool fullT = (m0 + 16 <= M) && (n0 + 32 <= N);
  v8f acc[2] = {{}, {}};
#if HAVE_GLB_ASYNC
  if (fullT && alA && alW && transB && ((K & 31) == 0)) {
    float* s0 = &Asm[wid][0][0];
    float* s1 = &Asm[wid][1][0];
    issue_a_async(A, lda, m0, 0, lane, s0);
    int nk = K >> 5;
    int buf = 0;
    for (int kk = 0; kk < nk; ++kk) {
      int k0 = kk << 5;
      if (kk + 1 < nk) {
        issue_a_async(A, lda, m0, k0 + 32, lane, buf ? s0 : s1);
        wait_async<4>();                 // previous 4 copies (current buf) done
      } else {
        wait_async<0>();
      }
      v16h a  = read_a_lds(buf ? s1 : s0, lane);
      v16h b0 = load_bT_v4(W, ldw, n0,      k0, lane);
      v16h b1 = load_bT_v4(W, ldw, n0 + 16, k0, lane);
      acc[0] = __builtin_amdgcn_wmma_f32_16x16x32_f16(false, a, false, b0, (short)0, acc[0], false, false);
      acc[1] = __builtin_amdgcn_wmma_f32_16x16x32_f16(false, a, false, b1, (short)0, acc[1], false, false);
      buf ^= 1;
    }
  } else
#endif
  {
    for (int k0 = 0; k0 < K; k0 += 32) {
      v16h a, b0, b1;
      bool fast = fullT && (k0 + 32 <= K) && alA && (!transB || alW);
      if (fast) {
        if (k0 + 32 < K)
          __builtin_prefetch(A + (long)(m0 + (lane & 15)) * lda + k0 + 32, 0, 3); // global_prefetch_b8
        a = load_a_v4(A, lda, m0, k0, lane);
        if (transB) {
          b0 = load_bT_v4(W, ldw, n0,      k0, lane);
          b1 = load_bT_v4(W, ldw, n0 + 16, k0, lane);
        } else {
          b0 = load_bN_fast(W, ldw, n0,      k0, lane);
          b1 = load_bN_fast(W, ldw, n0 + 16, k0, lane);
        }
      } else {
        a = load_a_g(A, lda, M, K, m0, k0, lane);
        if (transB) {
          b0 = load_bT_g(W, ldw, N, K, n0,      k0, lane);
          b1 = load_bT_g(W, ldw, N, K, n0 + 16, k0, lane);
        } else {
          b0 = load_bN_g(W, ldw, N, K, n0,      k0, lane);
          b1 = load_bN_g(W, ldw, N, K, n0 + 16, k0, lane);
        }
      }
      acc[0] = __builtin_amdgcn_wmma_f32_16x16x32_f16(false, a, false, b0, (short)0, acc[0], false, false);
      acc[1] = __builtin_amdgcn_wmma_f32_16x16x32_f16(false, a, false, b1, (short)0, acc[1], false, false);
    }
  }
#pragma unroll
  for (int half = 0; half < 2; ++half) {
    int n = n0 + half * 16 + (lane & 15);
#pragma unroll
    for (int r = 0; r < 8; ++r) {
      int m = m0 + r + ((lane < 16) ? 0 : 8);
      if (m < M && n < N) {
        float v = acc[half][r];
        if (bias)  v += bias[n];
        if (accum) v += C[(long)m * ldc + n];
        C[(long)m * ldc + n] = dev_act(v, act);
      }
    }
  }
}

static void gemm(hipStream_t s, const float* A, const float* W, const float* bias, float* C,
                 int M, int N, int K, int lda, int ldw, int ldc,
                 int act = ACT_NONE, int accum = 0, int transB = 1,
                 int batch = 1, long sA = 0, long sW = 0, long sC = 0) {
  int waves = ((M + 15) / 16) * ((N + 31) / 32);
  dim3 g((waves + 7) / 8, batch);
  gemm_wmma_k<<<g, 256, 0, s>>>(A, W, bias, C, M, N, K, lda, ldw, ldc,
                                sA, sW, sC, transB, accum, act);
}

// ---------------------------------------------------------------- flash attention (WMMA)
// qkv: [B, L, 384] (q | k | v, each 128 = 4 heads * 32). out: [B, L, 128].
__global__ void __launch_bounds__(256)
flash_attn_k(const float* __restrict__ qkv, float* __restrict__ out) {
  __shared__ _Float16 Pb[8][16][32];
  int lane = threadIdx.x & 31;
  int wid  = threadIdx.x >> 5;
  int wave = blockIdx.x * 8 + wid;          // exactly B*NH*(L/16) waves launched
  int tiles = LSEQ >> 4;
  int b  = wave / (NHEADS * tiles);
  int r0 = wave % (NHEADS * tiles);
  int hh = r0 / tiles;
  int qt = r0 % tiles;
  const float* base = qkv + (long)b * LSEQ * 384;
  v16h aq = load_a_v4(base + hh * HD, 384, qt * 16, 0, lane);
  float mrow[8], lrow[8];
  v8f o0 = {}, o1 = {};
#pragma unroll
  for (int r = 0; r < 8; ++r) { mrow[r] = -3.0e38f; lrow[r] = 0.f; }
  const float scale = 0.17677669529f;       // 1/sqrt(32)
  for (int j0 = 0; j0 < LSEQ; j0 += 32) {
    v16h kb0 = load_bT_v4(base + 128 + hh * HD, 384, j0,      0, lane);
    v16h kb1 = load_bT_v4(base + 128 + hh * HD, 384, j0 + 16, 0, lane);
    v8f s0 = {}, s1 = {};
    s0 = __builtin_amdgcn_wmma_f32_16x16x32_f16(false, aq, false, kb0, (short)0, s0, false, false);
    s1 = __builtin_amdgcn_wmma_f32_16x16x32_f16(false, aq, false, kb1, (short)0, s1, false, false);
#pragma unroll
    for (int r = 0; r < 8; ++r) {
      float v0 = s0[r] * scale, v1 = s1[r] * scale;
      float mx = fmaxf(v0, v1);
      for (int d = 1; d < 16; d <<= 1) mx = fmaxf(mx, __shfl_xor(mx, d, 32));
      float mnew = fmaxf(mrow[r], mx);
      float corr = __expf(mrow[r] - mnew);
      float p0 = __expf(v0 - mnew), p1 = __expf(v1 - mnew);
      float ssum = p0 + p1;
      for (int d = 1; d < 16; d <<= 1) ssum += __shfl_xor(ssum, d, 32);
      lrow[r] = lrow[r] * corr + ssum;
      mrow[r] = mnew;
      o0[r] *= corr; o1[r] *= corr;
      int mloc = r + ((lane < 16) ? 0 : 8);
      Pb[wid][mloc][lane & 15]        = (_Float16)p0;
      Pb[wid][mloc][16 + (lane & 15)] = (_Float16)p1;
    }
    __syncthreads();
    v16h ap;
    {
      int mm = lane & 15;
      int kb = (lane < 16) ? 0 : 16;
#pragma unroll
      for (int i = 0; i < 16; ++i) ap[i] = Pb[wid][mm][kb + i];
    }
    __syncthreads();
    const float* vb = base + (long)j0 * 384 + 256 + hh * HD;
    v16h vf0 = load_bN_fast(vb, 384, 0,  0, lane);
    v16h vf1 = load_bN_fast(vb, 384, 16, 0, lane);
    o0 = __builtin_amdgcn_wmma_f32_16x16x32_f16(false, ap, false, vf0, (short)0, o0, false, false);
    o1 = __builtin_amdgcn_wmma_f32_16x16x32_f16(false, ap, false, vf1, (short)0, o1, false, false);
  }
  int n = lane & 15;
#pragma unroll
  for (int r = 0; r < 8; ++r) {
    int m = qt * 16 + r + ((lane < 16) ? 0 : 8);
    float inv = 1.f / lrow[r];
    out[((long)b * LSEQ + m) * 128 + hh * HD + n]      = o0[r] * inv;
    out[((long)b * LSEQ + m) * 128 + hh * HD + 16 + n] = o1[r] * inv;
  }
}

// ---------------------------------------------------------------- layernorm (wave per row, C=128)
__global__ void __launch_bounds__(256)
layernorm_k(const float* __restrict__ x, const float* __restrict__ g,
            const float* __restrict__ b, float* __restrict__ y, int rows) {
  int lane = threadIdx.x & 31;
  int row  = blockIdx.x * 8 + (threadIdx.x >> 5);
  if (row >= rows) return;
  const float* xr = x + (long)row * 128;
  float v[4], s = 0.f;
#pragma unroll
  for (int i = 0; i < 4; ++i) { v[i] = xr[lane + 32 * i]; s += v[i]; }
  for (int d = 1; d < 32; d <<= 1) s += __shfl_xor(s, d, 32);
  float mean = s * (1.f / 128.f);
  float q = 0.f;
#pragma unroll
  for (int i = 0; i < 4; ++i) { float t = v[i] - mean; q += t * t; }
  for (int d = 1; d < 32; d <<= 1) q += __shfl_xor(q, d, 32);
  float inv = rsqrtf(q * (1.f / 128.f) + 1e-5f);
  float* yr = y + (long)row * 128;
#pragma unroll
  for (int i = 0; i < 4; ++i) {
    int c = lane + 32 * i;
    yr[c] = (v[i] - mean) * inv * g[c] + b[c];
  }
}

// ---------------------------------------------------------------- stem
__global__ void stem_conv_k(const float* __restrict__ x,
                            const float* dw, const float* db,
                            const float* fw, const float* fb,
                            const float* bg, const float* bb,
                            const float* brm, const float* brv,
                            float* __restrict__ feat) {
  int idx = blockIdx.x * blockDim.x + threadIdx.x;
  if (idx >= BL) return;
  int b = idx / LSEQ, t = idx % LSEQ;
  const float* xb = x + (long)b * LSEQ;
  float* fr = feat + (long)idx * 128;
  for (int c = 0; c < 64; ++c) {
    float acc = db[c];
    for (int j = 0; j < 3; ++j) {
      int tt = t + j - 1;
      float xd = 0.f;
      if (tt >= 0 && tt < LSEQ) xd = (tt == 0) ? (xb[1] - xb[0]) : (xb[tt] - xb[tt - 1]);
      acc += dw[c * 3 + j] * xd;
    }
    acc = (acc - brm[c]) * rsqrtf(brv[c] + 1e-5f) * bg[c] + bb[c];
    fr[c] = dev_act(acc, ACT_GELU);
  }
  for (int c = 0; c < 64; ++c) {
    float acc = fb[c];
    for (int j = 0; j < 7; ++j) {
      int tt = t + j - 3;
      acc += fw[c * 7 + j] * ((tt >= 0 && tt < LSEQ) ? xb[tt] : 0.f);
    }
    int cc = 64 + c;
    acc = (acc - brm[cc]) * rsqrtf(brv[cc] + 1e-5f) * bg[cc] + bb[cc];
    fr[cc] = dev_act(acc, ACT_GELU);
  }
}

__global__ void colmean_k(const float* __restrict__ feat, float* __restrict__ pooled) {
  int i = blockIdx.x * blockDim.x + threadIdx.x;
  if (i >= BSZ * 128) return;
  int c = i & 127, b = i >> 7;
  float acc = 0.f;
  for (int t = 0; t < LSEQ; ++t) acc += feat[((long)b * LSEQ + t) * 128 + c];
  pooled[i] = acc * (1.f / LSEQ);
}

__global__ void se_scale_k(const float* __restrict__ feat, const float* __restrict__ a,
                           float* __restrict__ h) {
  long i = (long)blockIdx.x * blockDim.x + threadIdx.x;
  if (i >= (long)BL * 128) return;
  int c = i & 127;
  long row = i >> 7;
  int b = (int)(row / LSEQ);
  h[i] = feat[i] * a[b * 128 + c];
}

// ---------------------------------------------------------------- mq helpers
__global__ void downsample_k(const float* __restrict__ h, float* __restrict__ dst, int s) {
  long i = (long)blockIdx.x * blockDim.x + threadIdx.x;
  long total = (long)BSZ * (LSEQ / s) * 128;
  if (i >= total) return;
  int c = i & 127;
  long row = i >> 7;
  int Ld = LSEQ / s;
  int td = (int)(row % Ld);
  long b = row / Ld;
  float acc = 0.f;
  for (int j = 0; j < s; ++j) acc += h[((long)b * LSEQ + td * s + j) * 128 + c];
  dst[i] = acc / (float)s;
}

__global__ void interp_k(const float* __restrict__ q, float* __restrict__ dst,
                         int Lh, int colOff) {
  long i = (long)blockIdx.x * blockDim.x + threadIdx.x;
  if (i >= (long)BL * 128) return;
  int c = i & 127;
  long row = i >> 7;
  int t = (int)(row % LSEQ);
  long b = row / LSEQ;
  float src = (t + 0.5f) * (float)Lh / (float)LSEQ - 0.5f;
  src = fminf(fmaxf(src, 0.f), (float)(Lh - 1));
  int i0 = (int)src;
  int i1 = (i0 + 1 < Lh) ? i0 + 1 : Lh - 1;
  float wg = src - (float)i0;
  float v = q[((long)b * Lh + i0) * 128 + c] * (1.f - wg) + q[((long)b * Lh + i1) * 128 + c] * wg;
  dst[row * 384 + colOff + c] = v;
}

// ---------------------------------------------------------------- mamba pieces
__global__ void mamba_conv_k(const float* __restrict__ xz, const float* cw,
                             const float* cb, float* __restrict__ u) {
  long i = (long)blockIdx.x * blockDim.x + threadIdx.x;
  if (i >= (long)BL * DINNER) return;
  int d = i & 255;
  long row = i >> 8;
  int t = (int)(row % LSEQ);
  long b = row / LSEQ;
  float acc = cb[d];
#pragma unroll
  for (int k = 0; k < 4; ++k) {
    int tt = t + k - 3;
    if (tt >= 0) acc += cw[d * 4 + k] * xz[((long)(b * LSEQ + tt)) * 512 + d];
  }
  u[i] = acc / (1.f + __expf(-acc));   // silu
}

__global__ void mamba_scan_k(const float* __restrict__ dt, const float* __restrict__ dbl,
                             const float* __restrict__ u, const float* __restrict__ A_log,
                             const float* __restrict__ Dp, float* __restrict__ y) {
  int tid = blockIdx.x * blockDim.x + threadIdx.x;
  if (tid >= BSZ * DINNER) return;
  int b = tid >> 8, d = tid & 255;
  float Av[DSTATE], h[DSTATE];
#pragma unroll
  for (int s = 0; s < DSTATE; ++s) { Av[s] = -__expf(A_log[d * DSTATE + s]); h[s] = 0.f; }
  float Dd = Dp[d];
  for (int t = 0; t < LSEQ; ++t) {
    long row = (long)b * LSEQ + t;
    float dtv = dt[row * 256 + d];
    float uv  = u[row * 256 + d];
    const float* db_ = dbl + row * 40;
    float acc = 0.f;
#pragma unroll
    for (int s = 0; s < DSTATE; ++s) {
      float dA = __expf(dtv * Av[s]);
      h[s] = dA * h[s] + dtv * db_[8 + s] * uv;
      acc += h[s] * db_[24 + s];
    }
    y[row * 256 + d] = acc + uv * Dd;
  }
}

__global__ void mamba_zgate_k(float* __restrict__ y, const float* __restrict__ xz) {
  long i = (long)blockIdx.x * blockDim.x + threadIdx.x;
  if (i >= (long)BL * DINNER) return;
  int d = i & 255;
  long row = i >> 8;
  float z = xz[row * 512 + 256 + d];
  y[i] *= z / (1.f + __expf(-z));
}

// ---------------------------------------------------------------- misc elementwise
__global__ void concat_k(const float* __restrict__ a, const float* __restrict__ b,
                         float* __restrict__ dst) {
  long i = (long)blockIdx.x * blockDim.x + threadIdx.x;
  if (i >= (long)BL * 256) return;
  int c = i & 255;
  long row = i >> 8;
  dst[i] = (c < 128) ? a[row * 128 + c] : b[row * 128 + c - 128];
}

__global__ void sssd_combine_k(float* __restrict__ x, const float* __restrict__ g,
                               const float* __restrict__ ssm, const float* __restrict__ attn) {
  long i = (long)blockIdx.x * blockDim.x + threadIdx.x;
  if (i >= (long)BL * 128) return;
  float gv = g[i];
  x[i] = x[i] + gv * ssm[i] + (1.f - gv) * attn[i];
}

__global__ void add2_k(float* __restrict__ x, const float* __restrict__ t, long total) {
  long i = (long)blockIdx.x * blockDim.x + threadIdx.x;
  if (i < total) x[i] += t[i];
}

__global__ void fill_fwd_dft_k(float* __restrict__ cm, float* __restrict__ sm) {
  long i = (long)blockIdx.x * blockDim.x + threadIdx.x;
  if (i >= (long)FREQ * LSEQ) return;
  int f = (int)(i / LSEQ), t = (int)(i % LSEQ);
  float th = 6.28318530718f * (float)(((long)f * t) & 1023) * (1.f / 1024.f);
  cm[i] = cosf(th);
  sm[i] = -sinf(th);
}

__global__ void fill_inv_dft_k(float* __restrict__ wr, float* __restrict__ wi) {
  long i = (long)blockIdx.x * blockDim.x + threadIdx.x;
  if (i >= (long)LSEQ * FREQP) return;
  int f = (int)(i % FREQP), t = (int)(i / FREQP);
  if (f >= FREQ) { wr[i] = 0.f; wi[i] = 0.f; return; }
  float th = 6.28318530718f * (float)(((long)f * t) & 1023) * (1.f / 1024.f);
  float cf = (f == 0 || f == 512) ? 1.f : 2.f;
  wr[i] = cf * cosf(th) * (1.f / 1024.f);
  wi[i] = -cf * sinf(th) * (1.f / 1024.f);
}

__global__ void head_k(const float* __restrict__ p1, const float* __restrict__ w,
                       const float* __restrict__ b, float* __restrict__ out) {
  int row = blockIdx.x * blockDim.x + threadIdx.x;
  if (row >= BL) return;
  float acc = b[0];
  for (int c = 0; c < 64; ++c) acc += p1[(long)row * 64 + c] * w[c];
  out[row] = acc;
}

// ---------------------------------------------------------------- launch
static inline unsigned g1(long n) { return (unsigned)((n + 255) / 256); }

extern "C" void kernel_launch(void* const* d_in, const int* in_sizes, int n_in,
                              void* d_out, int out_size, void* d_ws, size_t ws_size,
                              hipStream_t stream) {
  (void)in_sizes; (void)n_in; (void)out_size; (void)ws_size;
  auto in = [&](int i) { return (const float*)d_in[i]; };
  int P = 0;
  const float* x = in(P++);
  // stem (dict insertion order of the reference pytree)
  const float *st_dw = in(P++), *st_db = in(P++), *st_fw = in(P++), *st_fb = in(P++),
              *st_bg = in(P++), *st_bb = in(P++), *st_rm = in(P++), *st_rv = in(P++),
              *st_f1w = in(P++), *st_f1b = in(P++), *st_f2w = in(P++), *st_f2b = in(P++);
  // mq
  const float *mq_pw[3], *mq_pb[3];
  for (int i = 0; i < 3; ++i) mq_pw[i] = in(P++);
  for (int i = 0; i < 3; ++i) mq_pb[i] = in(P++);
  const float *mq_fw = in(P++), *mq_fb = in(P++);
  struct Blk {
    const float *ln_g, *ln_b, *in_proj, *conv_w, *conv_b, *x_proj, *dt_w, *dt_b,
                *A_log, *Dp, *out_proj, *a_in_w, *a_in_b, *a_out_w, *a_out_b,
                *gate_w, *gate_b;
  };
  auto readBlk = [&](Blk& B) {
    B.ln_g = in(P++); B.ln_b = in(P++);
    B.in_proj = in(P++); B.conv_w = in(P++); B.conv_b = in(P++); B.x_proj = in(P++);
    B.dt_w = in(P++); B.dt_b = in(P++); B.A_log = in(P++); B.Dp = in(P++); B.out_proj = in(P++);
    B.a_in_w = in(P++); B.a_in_b = in(P++); B.a_out_w = in(P++); B.a_out_b = in(P++);
    B.gate_w = in(P++); B.gate_b = in(P++);
  };
  Blk sa_b[3], ca_b[3];
  for (int i = 0; i < 3; ++i) readBlk(sa_b[i]);
  const float *ca_ng = in(P++), *ca_nb = in(P++);
  const float *caa_in_w = in(P++), *caa_in_b = in(P++), *caa_out_w = in(P++), *caa_out_b = in(P++);
  for (int i = 0; i < 3; ++i) readBlk(ca_b[i]);
  struct Fdf { const float *ln_g, *ln_b, *w1, *b1, *w2, *b2, *wt, *bt; };
  auto readFdf = [&](Fdf& F) {
    F.ln_g = in(P++); F.ln_b = in(P++); F.w1 = in(P++); F.b1 = in(P++);
    F.w2 = in(P++); F.b2 = in(P++); F.wt = in(P++); F.bt = in(P++);
  };
  Fdf fdf_sa, fdf_ca;
  readFdf(fdf_sa); readFdf(fdf_ca);
  const float *fus_w = in(P++), *fus_b = in(P++), *fn_g = in(P++), *fn_b = in(P++),
              *p1_w = in(P++), *p1_b = in(P++), *p2_w = in(P++), *p2_b = in(P++);

  // workspace carve (bump allocator)
  char* w8 = (char*)d_ws;
  size_t off = 0;
  auto alloc = [&](long nfloats) {
    float* p = (float*)(w8 + off);
    off += (((size_t)nfloats * 4 + 255) & ~(size_t)255);
    return p;
  };
  float* t_h    = alloc((long)BL * 128);
  float* t_can  = alloc((long)BL * 128);
  float* t_query= alloc((long)BL * 128);
  float* t_sa   = alloc((long)BL * 128);
  float* t_ca   = alloc((long)BL * 128);
  float* t_xn   = alloc((long)BL * 128);
  float* t_ssm  = alloc((long)BL * 128);
  float* t_attn = alloc((long)BL * 128);
  float* t_fo   = alloc((long)BL * 128);
  float* t_gate = alloc((long)BL * 128);
  float* t_xt   = alloc((long)BL * 128);
  float* t_feat = alloc((long)BL * 128);
  float* t_xz   = alloc((long)BL * 512);
  float* t_u    = alloc((long)BL * 256);
  float* t_dt   = alloc((long)BL * 256);
  float* t_y    = alloc((long)BL * 256);
  float* t_cat  = alloc((long)BL * 256);
  float* t_dbl  = alloc((long)BL * 40);
  float* t_qkv  = alloc((long)BL * 384);
  float* t_MR   = alloc((long)BSZ * FREQ * 128);
  float* t_MI   = alloc((long)BSZ * FREQ * 128);
  float* t_OR   = alloc((long)BSZ * FREQ * 128);
  float* t_OI   = alloc((long)BSZ * FREQ * 128);
  float* t_cos  = alloc((long)FREQ * LSEQ);
  float* t_sin  = alloc((long)FREQ * LSEQ);
  float* t_wr   = alloc((long)LSEQ * FREQP);
  float* t_wi   = alloc((long)LSEQ * FREQP);
  float* t_xd2  = alloc((long)BSZ * 512 * 128);
  float* t_q2   = alloc((long)BSZ * 512 * 128);
  float* t_xd4  = alloc((long)BSZ * 256 * 128);
  float* t_q4   = alloc((long)BSZ * 256 * 128);
  float* t_pool = alloc(BSZ * 128);
  float* t_se1  = alloc(BSZ * 32);
  float* t_sea  = alloc(BSZ * 128);
  float* t_p1   = alloc((long)BL * 64);
  float* t_hid  = t_xz;   // alias: FDF hidden [B*F,512] fits in xz region (idle then)

  // DFT matrices
  fill_fwd_dft_k<<<g1((long)FREQ * LSEQ), 256, 0, stream>>>(t_cos, t_sin);
  fill_inv_dft_k<<<g1((long)LSEQ * FREQP), 256, 0, stream>>>(t_wr, t_wi);

  // ---------- stem ----------
  stem_conv_k<<<g1(BL), 256, 0, stream>>>(x, st_dw, st_db, st_fw, st_fb,
                                          st_bg, st_bb, st_rm, st_rv, t_feat);
  colmean_k<<<g1(BSZ * 128), 256, 0, stream>>>(t_feat, t_pool);
  gemm(stream, t_pool, st_f1w, st_f1b, t_se1, BSZ, 32, 128, 128, 128, 32, ACT_RELU);
  gemm(stream, t_se1, st_f2w, st_f2b, t_sea, BSZ, 128, 32, 32, 32, 128, ACT_SIGMOID);
  se_scale_k<<<g1((long)BL * 128), 256, 0, stream>>>(t_feat, t_sea, t_h);

  // ---------- multi-scale query ----------
  gemm(stream, t_h, mq_pw[0], mq_pb[0], t_qkv, BL, 128, 128, 128, 128, 384);
  downsample_k<<<g1((long)BSZ * 512 * 128), 256, 0, stream>>>(t_h, t_xd2, 2);
  gemm(stream, t_xd2, mq_pw[1], mq_pb[1], t_q2, BSZ * 512, 128, 128, 128, 128, 128);
  interp_k<<<g1((long)BL * 128), 256, 0, stream>>>(t_q2, t_qkv, 512, 128);
  downsample_k<<<g1((long)BSZ * 256 * 128), 256, 0, stream>>>(t_h, t_xd4, 4);
  gemm(stream, t_xd4, mq_pw[2], mq_pb[2], t_q4, BSZ * 256, 128, 128, 128, 128, 128);
  interp_k<<<g1((long)BL * 128), 256, 0, stream>>>(t_q4, t_qkv, 256, 256);
  gemm(stream, t_qkv, mq_fw, mq_fb, t_query, BL, 128, 384, 384, 384, 128);

  // ---------- SSSD block runner ----------
  auto run_sssd = [&](float* xbuf, const Blk& B) {
    layernorm_k<<<(BL + 7) / 8, 256, 0, stream>>>(xbuf, B.ln_g, B.ln_b, t_xn, BL);
    // mamba
    gemm(stream, t_xn, B.in_proj, nullptr, t_xz, BL, 512, 128, 128, 128, 512);
    mamba_conv_k<<<g1((long)BL * 256), 256, 0, stream>>>(t_xz, B.conv_w, B.conv_b, t_u);
    gemm(stream, t_u, B.x_proj, nullptr, t_dbl, BL, 40, 256, 256, 256, 40);
    gemm(stream, t_dbl, B.dt_w, B.dt_b, t_dt, BL, 256, DTRANK, 40, DTRANK, 256, ACT_SOFTPLUS);
    mamba_scan_k<<<(BSZ * DINNER + 255) / 256, 256, 0, stream>>>(t_dt, t_dbl, t_u, B.A_log, B.Dp, t_y);
    mamba_zgate_k<<<g1((long)BL * 256), 256, 0, stream>>>(t_y, t_xz);
    gemm(stream, t_y, B.out_proj, nullptr, t_ssm, BL, 128, 256, 256, 256, 128);
    // attention
    gemm(stream, t_xn, B.a_in_w, B.a_in_b, t_qkv, BL, 384, 128, 128, 128, 384);
    flash_attn_k<<<BSZ * NHEADS * (LSEQ / 16) / 8, 256, 0, stream>>>(t_qkv, t_fo);
    gemm(stream, t_fo, B.a_out_w, B.a_out_b, t_attn, BL, 128, 128, 128, 128, 128);
    // gate + residual
    concat_k<<<g1((long)BL * 256), 256, 0, stream>>>(t_ssm, t_attn, t_cat);
    gemm(stream, t_cat, B.gate_w, B.gate_b, t_gate, BL, 128, 256, 256, 256, 128, ACT_SIGMOID);
    sssd_combine_k<<<g1((long)BL * 128), 256, 0, stream>>>(xbuf, t_gate, t_ssm, t_attn);
  };

  // ---------- FDF runner (DFT as GEMM) ----------
  auto run_fdf = [&](float* xbuf, const Fdf& F) {
    layernorm_k<<<(BL + 7) / 8, 256, 0, stream>>>(xbuf, F.ln_g, F.ln_b, t_xn, BL);
    gemm(stream, t_cos, t_xn, nullptr, t_MR, FREQ, 128, LSEQ, LSEQ, 128, 128,
         ACT_NONE, 0, /*transB*/0, BSZ, 0, (long)LSEQ * 128, (long)FREQ * 128);
    gemm(stream, t_sin, t_xn, nullptr, t_MI, FREQ, 128, LSEQ, LSEQ, 128, 128,
         ACT_NONE, 0, 0, BSZ, 0, (long)LSEQ * 128, (long)FREQ * 128);
    gemm(stream, t_MR, F.w1, F.b1, t_hid, BSZ * FREQ, 512, 128, 128, 128, 512, ACT_GELU);
    gemm(stream, t_hid, F.w2, F.b2, t_OR, BSZ * FREQ, 128, 512, 512, 512, 128);
    gemm(stream, t_MI, F.w1, F.b1, t_hid, BSZ * FREQ, 512, 128, 128, 128, 512, ACT_GELU);
    gemm(stream, t_hid, F.w2, F.b2, t_OI, BSZ * FREQ, 128, 512, 512, 512, 128);
    gemm(stream, t_wr, t_OR, nullptr, t_xt, LSEQ, 128, FREQP, FREQP, 128, 128,
         ACT_NONE, 0, 0, BSZ, 0, (long)FREQ * 128, (long)LSEQ * 128);
    gemm(stream, t_wi, t_OI, nullptr, t_xt, LSEQ, 128, FREQP, FREQP, 128, 128,
         ACT_NONE, 1, 0, BSZ, 0, (long)FREQ * 128, (long)LSEQ * 128);
    gemm(stream, t_xn, F.wt, F.bt, t_xt, BL, 128, 128, 128, 128, 128, ACT_NONE, 1);
    add2_k<<<g1((long)BL * 128), 256, 0, stream>>>(xbuf, t_xt, (long)BL * 128);
  };

  // ---------- self path ----------
  (void)hipMemcpyAsync(t_sa, t_h, (size_t)BL * 128 * 4, hipMemcpyDeviceToDevice, stream);
  for (int i = 0; i < 3; ++i) run_sssd(t_sa, sa_b[i]);
  run_fdf(t_sa, fdf_sa);

  // ---------- cross path ----------
  layernorm_k<<<(BL + 7) / 8, 256, 0, stream>>>(t_h, ca_ng, ca_nb, t_can, BL);
  gemm(stream, t_query, caa_in_w, caa_in_b, t_qkv, BL, 128, 128, 128, 128, 384);
  gemm(stream, t_can, caa_in_w + 128 * 128, caa_in_b + 128, t_qkv + 128,
       BL, 256, 128, 128, 128, 384);
  flash_attn_k<<<BSZ * NHEADS * (LSEQ / 16) / 8, 256, 0, stream>>>(t_qkv, t_fo);
  (void)hipMemcpyAsync(t_ca, t_h, (size_t)BL * 128 * 4, hipMemcpyDeviceToDevice, stream);
  gemm(stream, t_fo, caa_out_w, caa_out_b, t_ca, BL, 128, 128, 128, 128, 128, ACT_NONE, 1);
  for (int i = 0; i < 3; ++i) run_sssd(t_ca, ca_b[i]);
  run_fdf(t_ca, fdf_ca);

  // ---------- fusion head ----------
  concat_k<<<g1((long)BL * 256), 256, 0, stream>>>(t_sa, t_ca, t_cat);
  gemm(stream, t_cat, fus_w, fus_b, t_ssm, BL, 128, 256, 256, 256, 128);
  layernorm_k<<<(BL + 7) / 8, 256, 0, stream>>>(t_ssm, fn_g, fn_b, t_attn, BL);
  gemm(stream, t_attn, p1_w, p1_b, t_p1, BL, 64, 128, 128, 128, 64, ACT_GELU);
  head_k<<<g1(BL), 256, 0, stream>>>(t_p1, p2_w, p2_b, (float*)d_out);
}
